// MHAPoolCell_64372969833014
// MI455X (gfx1250) — compile-verified, matrix operand order checked by reading, not run on hardware
//
#include <hip/hip_runtime.h>
#include <hip/hip_bf16.h>

typedef __attribute__((ext_vector_type(16))) _Float16 v16h;
typedef __attribute__((ext_vector_type(8)))  _Float16 v8h;
typedef __attribute__((ext_vector_type(8)))  float    v8f;
typedef __attribute__((ext_vector_type(4)))  unsigned int u32x4;
typedef __attribute__((ext_vector_type(8)))  int      i32x8;
typedef __attribute__((ext_vector_type(4)))  int      i32x4;

#define B_  4
#define S_  2048
#define D_  512
#define H_  8
#define HD_ 64

#if defined(__has_builtin)
#if __has_builtin(__builtin_amdgcn_tensor_load_to_lds) && \
    __has_builtin(__builtin_amdgcn_s_wait_tensorcnt)
#define USE_TDM 1
#endif
#endif
#ifndef USE_TDM
#define USE_TDM 0
#endif

// ---------------------------------------------------------------------------
// Fragment loaders. Row-per-lane contiguous memory (A row-major, W
// pre-transposed to [N,K]) matching CDNA5 16-bit A 16x32 VGPR layout:
// lanes 0-15 hold K=0..7 / 16..23 of row (lane&15); lanes 16-31 hold
// K=8..15 / 24..31.
// ---------------------------------------------------------------------------
static __device__ inline v16h load_frag(const _Float16* __restrict__ base,
                                        int ld, int row0, int k0) {
  const int lane = threadIdx.x & 31;
  const int half = lane >> 4;
  const int r    = lane & 15;
  const _Float16* p = base + (size_t)(row0 + r) * ld + k0 + half * 8;
  v8h lo = *(const v8h*)(p);
  v8h hi = *(const v8h*)(p + 16);
  v16h out;
#pragma unroll
  for (int i = 0; i < 8; ++i) { out[i] = lo[i]; out[i + 8] = hi[i]; }
  return out;
}

// LDS tile fragment with parametric row pitch (f16 elements) and K offset.
static __device__ inline v16h lds_frag_p(const _Float16* base, int pitch,
                                         int r, int half, int k0) {
  const _Float16* p = base + r * pitch + k0 + half * 8;
  v8h lo = *(const v8h*)(p);
  v8h hi = *(const v8h*)(p + 16);
  v16h out;
#pragma unroll
  for (int i = 0; i < 8; ++i) { out[i] = lo[i]; out[i + 8] = hi[i]; }
  return out;
}

static __device__ inline v8f wmma_f16(v16h a, v16h b, v8f c) {
  return __builtin_amdgcn_wmma_f32_16x16x32_f16(
      /*neg_a=*/false, a, /*neg_b=*/false, b,
      /*c_mod=*/(short)0, c, /*reuse_a=*/false, /*reuse_b=*/false);
}

#if USE_TDM
// ---------------------------------------------------------------------------
// Tensor Data Mover: DMA a tile_d1 x tile_d0 f16 tile (row stride `ld_elems`)
// from global memory into LDS (rows packed). D# per ISA 08 §8.3/8.4.
// ---------------------------------------------------------------------------
static __device__ inline void tdm_load_tile(unsigned lds_byte_off,
                                            const _Float16* gptr,
                                            unsigned ld_elems,
                                            unsigned tile_d0,
                                            unsigned tile_d1) {
  unsigned long long ga = (unsigned long long)gptr;
  unsigned ga_lo = (unsigned)__builtin_amdgcn_readfirstlane((int)(unsigned)(ga & 0xFFFFFFFFull));
  unsigned ga_hi = (unsigned)__builtin_amdgcn_readfirstlane((int)(unsigned)(ga >> 32));
  unsigned lds   = (unsigned)__builtin_amdgcn_readfirstlane((int)lds_byte_off);

  u32x4 g0;
  g0[0] = 1u;                                   // count=1, user mode, no gather
  g0[1] = lds;                                  // lds_addr [63:32]
  g0[2] = ga_lo;                                // global_addr [95:64]
  g0[3] = (ga_hi & 0x01FFFFFFu) | (2u << 30);   // global_addr hi (57b), type=2

  const unsigned td0 = ld_elems;                // tensor_dim0 (elements)
  const unsigned td1 = 1u << 20;                // tensor_dim1 (rows, OOB bound)
  i32x8 g1;
  g1[0] = (int)(1u << 16);                      // workgroup_mask=0, data_size=2B
  g1[1] = (int)((td0 & 0xFFFFu) << 16);         // tensor_dim0 [63:48]
  g1[2] = (int)(((td0 >> 16) & 0xFFFFu) | ((td1 & 0xFFFFu) << 16));
  g1[3] = (int)(((td1 >> 16) & 0xFFFFu) | ((tile_d0 & 0xFFFFu) << 16));
  g1[4] = (int)(tile_d1 & 0xFFFFu);             // tile_dim1; tile_dim2=0
  g1[5] = (int)ld_elems;                        // tensor_dim0_stride lo32
  g1[6] = 0;                                    // stride0 hi, stride1 lo
  g1[7] = 0;                                    // stride1 hi
  i32x4 z4 = {0, 0, 0, 0};
#if __clang_major__ >= 23
  i32x8 z8 = {0, 0, 0, 0, 0, 0, 0, 0};
  __builtin_amdgcn_tensor_load_to_lds(g0, g1, z4, z4, z8, 0);
#else
  __builtin_amdgcn_tensor_load_to_lds(g0, g1, z4, z4, 0);
#endif
}
#endif  // USE_TDM

// ---------------------------------------------------------------------------
// Weight convert + transpose: W[K,N] f32 -> WT[N,K] f16
// ---------------------------------------------------------------------------
__global__ void wT_cvt(const float* __restrict__ W, _Float16* __restrict__ WT) {
  int idx = blockIdx.x * 256 + threadIdx.x;   // 512*512
  int k = idx >> 9, n = idx & 511;
  WT[(size_t)n * 512 + k] = (_Float16)W[(size_t)k * 512 + n];
}

__global__ void b_cvt(const float* __restrict__ b, _Float16* __restrict__ o) {
  int i = blockIdx.x * 256 + threadIdx.x;
  if (i < 512) o[i] = (_Float16)b[i];
}

// ---------------------------------------------------------------------------
// L2 normalize rows of x[B*S, D] (f32) -> f16
// ---------------------------------------------------------------------------
__global__ __launch_bounds__(256) void l2norm_cvt(const float* __restrict__ x,
                                                  _Float16* __restrict__ out) {
  const int wave = threadIdx.x >> 5, lane = threadIdx.x & 31;
  const int row  = blockIdx.x * 8 + wave;
  const float* p = x + (size_t)row * D_;
  float ss = 0.f, vals[16];
#pragma unroll
  for (int i = 0; i < 16; ++i) { float v = p[lane * 16 + i]; vals[i] = v; ss += v * v; }
#pragma unroll
  for (int off = 16; off >= 1; off >>= 1) ss += __shfl_xor(ss, off, 32);
  float inv = 1.0f / fmaxf(sqrtf(ss), 1e-12f);
  _Float16* o = out + (size_t)row * D_;
#pragma unroll
  for (int i = 0; i < 16; ++i) o[lane * 16 + i] = (_Float16)(vals[i] * inv);
}

// ---------------------------------------------------------------------------
// GEMM: C[8192,512] = A[8192,512] (f16) x WT[512,512] (f16,[N,K]) + bias,
// optional ReLU, f16 out. Block tile 128x128, 8 waves (2x4), per-wave 64x32
// => 4x2 accumulators, 8 WMMA per 32-wide K step.
// TDM path: both 128x32 panels staged in LDS by the Tensor Data Mover,
// double-buffered; wave 0 issues DMAs and gates on s_wait_tensorcnt.
// ---------------------------------------------------------------------------
__global__ __launch_bounds__(256) void gemm_f16(const _Float16* __restrict__ A,
                                                const _Float16* __restrict__ WT,
                                                const _Float16* __restrict__ bias,
                                                _Float16* __restrict__ C,
                                                int relu) {
  const int wave = threadIdx.x >> 5;
  const int lane = threadIdx.x & 31;
  const int half = lane >> 4;
  const int r    = lane & 15;
  const int bm = blockIdx.x * 128;
  const int bn = blockIdx.y * 128;
  const int wm = (wave >> 2) * 64;
  const int wn = (wave & 3) * 32;

  v8f acc[4][2] = {};

#if USE_TDM
  __shared__ __align__(16) _Float16 ldsA[2][128 * 32];
  __shared__ __align__(16) _Float16 ldsB[2][128 * 32];
  if (wave == 0) {
    tdm_load_tile((unsigned)(size_t)&ldsA[0][0], A  + (size_t)bm * D_,      D_, 32, 128);
    tdm_load_tile((unsigned)(size_t)&ldsB[0][0], WT + (size_t)bn * D_,      D_, 32, 128);
    tdm_load_tile((unsigned)(size_t)&ldsA[1][0], A  + (size_t)bm * D_ + 32, D_, 32, 128);
    tdm_load_tile((unsigned)(size_t)&ldsB[1][0], WT + (size_t)bn * D_ + 32, D_, 32, 128);
  }
#pragma unroll 1
  for (int kt = 0; kt < D_ / 32; ++kt) {
    const int buf = kt & 1;
    if (wave == 0) __builtin_amdgcn_s_wait_tensorcnt(2);  // tiles for kt landed
    __syncthreads();
    v16h b0 = lds_frag_p(&ldsB[buf][(wn)      * 32], 32, r, half, 0);
    v16h b1 = lds_frag_p(&ldsB[buf][(wn + 16) * 32], 32, r, half, 0);
#pragma unroll
    for (int i = 0; i < 4; ++i) {
      v16h a = lds_frag_p(&ldsA[buf][(wm + 16 * i) * 32], 32, r, half, 0);
      acc[i][0] = wmma_f16(a, b0, acc[i][0]);
      acc[i][1] = wmma_f16(a, b1, acc[i][1]);
    }
    __syncthreads();  // everyone done with buf before DMA overwrites it
    if (wave == 0 && kt + 2 < D_ / 32) {
      tdm_load_tile((unsigned)(size_t)&ldsA[buf][0], A  + (size_t)bm * D_ + (kt + 2) * 32, D_, 32, 128);
      tdm_load_tile((unsigned)(size_t)&ldsB[buf][0], WT + (size_t)bn * D_ + (kt + 2) * 32, D_, 32, 128);
    }
  }
  if (wave == 0) __builtin_amdgcn_s_wait_tensorcnt(0);
#else
#pragma unroll 1
  for (int k0 = 0; k0 < D_; k0 += 32) {
    if (k0 + 32 < D_) {
      __builtin_prefetch(A  + (size_t)(bm + wm + r) * D_ + k0 + 32, 0, 1);
      __builtin_prefetch(WT + (size_t)(bn + wn + r) * D_ + k0 + 32, 0, 1);
    }
    v16h b0 = load_frag(WT, D_, bn + wn,      k0);
    v16h b1 = load_frag(WT, D_, bn + wn + 16, k0);
#pragma unroll
    for (int i = 0; i < 4; ++i) {
      v16h a = load_frag(A, D_, bm + wm + 16 * i, k0);
      acc[i][0] = wmma_f16(a, b0, acc[i][0]);
      acc[i][1] = wmma_f16(a, b1, acc[i][1]);
    }
  }
#endif

#pragma unroll
  for (int i = 0; i < 4; ++i)
#pragma unroll
    for (int j = 0; j < 2; ++j) {
      const int col = bn + wn + j * 16 + r;
      const float bv = (float)bias[col];
#pragma unroll
      for (int v = 0; v < 8; ++v) {
        const int row = bm + wm + 16 * i + v + 8 * half;
        float x = acc[i][j][v] + bv;
        if (relu) x = fmaxf(x, 0.f);
        C[(size_t)row * D_ + col] = (_Float16)x;
      }
    }
}

// ---------------------------------------------------------------------------
// Per-head transpose: v[b*S+s, h*HD+hd] -> vT[((b*H+h)*HD+hd)*S + s]
// ---------------------------------------------------------------------------
__global__ void head_transpose(const _Float16* __restrict__ v,
                               _Float16* __restrict__ vT) {
  int idx = blockIdx.x * 256 + threadIdx.x;   // B*S*D
  int sg = idx >> 9;       // b*S+s
  int d  = idx & 511;
  int b  = sg >> 11;
  int s  = sg & 2047;
  int h  = d >> 6, hd = d & 63;
  vT[(((size_t)(b * H_ + h) * HD_ + hd) << 11) + s] = v[idx];
}

// ---------------------------------------------------------------------------
// Flash attention: one wave per (b, h, 16-query tile). 32-key blocks:
// scores via 4 WMMA (K=64), online softmax (shfl_xor row reductions in
// 16-lane halves), P restaged through LDS (C-layout -> A-layout), PV via
// 4 WMMA. TDM path: K-tile (32x64) and VT-tile (64x32) are DMAed into LDS,
// double-buffered; the single wave self-gates on s_wait_tensorcnt (no
// barrier needed), overlapping DMA with the softmax VALU work.
// ---------------------------------------------------------------------------
__global__ __launch_bounds__(32) void attn_fa(const _Float16* __restrict__ Q,
                                              const _Float16* __restrict__ Km,
                                              const _Float16* __restrict__ VT,
                                              const float* __restrict__ mask,
                                              _Float16* __restrict__ O) {
  const int lane = threadIdx.x & 31;
  const int half = lane >> 4;
  const int r    = lane & 15;
  const int q0 = blockIdx.x * 16;
  const int h  = blockIdx.y;
  const int b  = blockIdx.z;

  const _Float16* Qb = Q  + (size_t)b * S_ * D_ + h * HD_;
  const _Float16* Kb = Km + (size_t)b * S_ * D_ + h * HD_;
  const _Float16* Vb = VT + (size_t)(b * H_ + h) * HD_ * S_;
  const float*    mb = mask + (size_t)b * S_;

  const v16h qa0 = load_frag(Qb, D_, q0, 0);
  const v16h qa1 = load_frag(Qb, D_, q0, 32);

  float mrow[8], lrow[8];
  v8f oacc[4] = {};
#pragma unroll
  for (int v = 0; v < 8; ++v) { mrow[v] = -3.0e38f; lrow[v] = 0.f; }

#if USE_TDM
  __shared__ __align__(16) _Float16 klds[2][32 * 64];  // keys x hd
  __shared__ __align__(16) _Float16 vlds[2][64 * 32];  // hd x keys
#endif
  __shared__ __align__(16) _Float16 plds[16 * 32];

#if USE_TDM
  // prologue: tiles for kb=0 and kb=32
  tdm_load_tile((unsigned)(size_t)&klds[0][0], Kb,                D_, 64, 32);
  tdm_load_tile((unsigned)(size_t)&vlds[0][0], Vb,                S_, 32, 64);
  tdm_load_tile((unsigned)(size_t)&klds[1][0], Kb + (size_t)32 * D_, D_, 64, 32);
  tdm_load_tile((unsigned)(size_t)&vlds[1][0], Vb + 32,           S_, 32, 64);
#endif

#pragma unroll 1
  for (int kb = 0; kb < S_; kb += 32) {
    v8f s0 = {}, s1 = {};
#if USE_TDM
    const int buf = (kb >> 5) & 1;
    __builtin_amdgcn_s_wait_tensorcnt(2);   // K/V tiles for kb landed
    {
      v16h k0a = lds_frag_p(&klds[buf][0],       64, r, half, 0);
      v16h k0b = lds_frag_p(&klds[buf][0],       64, r, half, 32);
      s0 = wmma_f16(qa0, k0a, s0);
      s0 = wmma_f16(qa1, k0b, s0);
      v16h k1a = lds_frag_p(&klds[buf][16 * 64], 64, r, half, 0);
      v16h k1b = lds_frag_p(&klds[buf][16 * 64], 64, r, half, 32);
      s1 = wmma_f16(qa0, k1a, s1);
      s1 = wmma_f16(qa1, k1b, s1);
    }
#else
    {
      v16h k0a = load_frag(Kb, D_, kb, 0);
      v16h k0b = load_frag(Kb, D_, kb, 32);
      s0 = wmma_f16(qa0, k0a, s0);
      s0 = wmma_f16(qa1, k0b, s0);
      v16h k1a = load_frag(Kb, D_, kb + 16, 0);
      v16h k1b = load_frag(Kb, D_, kb + 16, 32);
      s1 = wmma_f16(qa0, k1a, s1);
      s1 = wmma_f16(qa1, k1b, s1);
    }
#endif
    const float mv0 = mb[kb + r] * 1.0e30f;
    const float mv1 = mb[kb + 16 + r] * 1.0e30f;
#pragma unroll
    for (int v = 0; v < 8; ++v) {
      float x0 = s0[v] * 0.125f - mv0;   // 1/sqrt(HD)=0.125, key-padding mask
      float x1 = s1[v] * 0.125f - mv1;
      float mx = fmaxf(x0, x1);
#pragma unroll
      for (int off = 8; off >= 1; off >>= 1) mx = fmaxf(mx, __shfl_xor(mx, off, 32));
      float mnew = fmaxf(mrow[v], mx);
      float corr = __expf(mrow[v] - mnew);
      float p0 = __expf(x0 - mnew);
      float p1 = __expf(x1 - mnew);
      float rs = p0 + p1;
#pragma unroll
      for (int off = 8; off >= 1; off >>= 1) rs += __shfl_xor(rs, off, 32);
      lrow[v] = lrow[v] * corr + rs;
      mrow[v] = mnew;
#pragma unroll
      for (int t = 0; t < 4; ++t) oacc[t][v] *= corr;
      const int prow = v + 8 * half;
      plds[prow * 32 + r]      = (_Float16)p0;
      plds[prow * 32 + 16 + r] = (_Float16)p1;
    }
    __syncthreads();
    v16h pa;
    {
      const _Float16* pp = &plds[r * 32 + half * 8];
      v8h lo = *(const v8h*)(pp);
      v8h hi = *(const v8h*)(pp + 16);
#pragma unroll
      for (int i = 0; i < 8; ++i) { pa[i] = lo[i]; pa[i + 8] = hi[i]; }
    }
#if USE_TDM
#pragma unroll
    for (int t = 0; t < 4; ++t) {
      v16h vf = lds_frag_p(&vlds[buf][t * 16 * 32], 32, r, half, 0);
      oacc[t] = wmma_f16(pa, vf, oacc[t]);
    }
    // issue tiles for kb+64 into this buffer (all ds reads of buf are done)
    if (kb + 64 < S_) {
      tdm_load_tile((unsigned)(size_t)&klds[buf][0], Kb + (size_t)(kb + 64) * D_, D_, 64, 32);
      tdm_load_tile((unsigned)(size_t)&vlds[buf][0], Vb + kb + 64,                S_, 32, 64);
    }
#else
#pragma unroll
    for (int t = 0; t < 4; ++t) {
      v16h vf = load_frag(Vb, S_, t * 16, kb);
      oacc[t] = wmma_f16(pa, vf, oacc[t]);
    }
#endif
    __syncthreads();
  }
#if USE_TDM
  __builtin_amdgcn_s_wait_tensorcnt(0);
#endif
#pragma unroll
  for (int v = 0; v < 8; ++v) {
    const float inv = 1.0f / lrow[v];
    const int row = q0 + v + 8 * half;
#pragma unroll
    for (int t = 0; t < 4; ++t)
      O[(size_t)(b * S_ + row) * D_ + h * HD_ + t * 16 + r] =
          (_Float16)(oacc[t][v] * inv);
  }
}

// ---------------------------------------------------------------------------
// LayerNorm rows of x[B*S, D] f16 -> f16
// ---------------------------------------------------------------------------
__global__ __launch_bounds__(256) void layernorm_f16(const _Float16* __restrict__ x,
                                                     const float* __restrict__ g,
                                                     const float* __restrict__ bb,
                                                     _Float16* __restrict__ out) {
  const int wave = threadIdx.x >> 5, lane = threadIdx.x & 31;
  const int row  = blockIdx.x * 8 + wave;
  const _Float16* p = x + (size_t)row * D_;
  float s = 0.f, ss = 0.f, vals[16];
#pragma unroll
  for (int i = 0; i < 16; ++i) { float v = (float)p[lane * 16 + i]; vals[i] = v; s += v; ss += v * v; }
#pragma unroll
  for (int off = 16; off >= 1; off >>= 1) { s += __shfl_xor(s, off, 32); ss += __shfl_xor(ss, off, 32); }
  float mu = s * (1.0f / D_);
  float var = ss * (1.0f / D_) - mu * mu;
  float rinv = rsqrtf(var + 1e-5f);
  _Float16* o = out + (size_t)row * D_;
#pragma unroll
  for (int i = 0; i < 16; ++i) {
    int d = lane * 16 + i;
    o[d] = (_Float16)((vals[i] - mu) * rinv * g[d] + bb[d]);
  }
}

// ---------------------------------------------------------------------------
// Pooling
// ---------------------------------------------------------------------------
__global__ __launch_bounds__(256) void pool_alpha_k(const _Float16* __restrict__ hbuf,
                                                    const float* __restrict__ pw,
                                                    const float* __restrict__ pb,
                                                    const float* __restrict__ mask,
                                                    float* __restrict__ alpha) {
  const int wave = threadIdx.x >> 5, lane = threadIdx.x & 31;
  const int row  = blockIdx.x * 8 + wave;   // b*S+s
  const _Float16* p = hbuf + (size_t)row * D_;
  float acc[8] = {};
  for (int i = 0; i < 16; ++i) {
    int d = lane * 16 + i;
    float x = (float)p[d];
#pragma unroll
    for (int hh = 0; hh < 8; ++hh) acc[hh] += x * pw[d * 8 + hh];
  }
#pragma unroll
  for (int hh = 0; hh < 8; ++hh)
#pragma unroll
    for (int off = 16; off >= 1; off >>= 1) acc[hh] += __shfl_xor(acc[hh], off, 32);
  if (lane == 0) {
    float mv = mask[row] * -1.0e16f;
#pragma unroll
    for (int hh = 0; hh < 8; ++hh) {
      float a = acc[hh] + pb[hh];
      a = a > 0.f ? a : 0.01f * a;                 // leaky_relu(0.01)
      alpha[(size_t)row * 8 + hh] = a + mv;
    }
  }
}

__global__ __launch_bounds__(256) void pool_stats_k(const float* __restrict__ alpha,
                                                    float* __restrict__ mbuf,
                                                    float* __restrict__ lbuf) {
  const int bh = blockIdx.x, b = bh >> 3, hh = bh & 7;
  __shared__ float red[256];
  float lm = -3.0e38f;
  for (int s = threadIdx.x; s < S_; s += 256)
    lm = fmaxf(lm, alpha[(size_t)(b * S_ + s) * 8 + hh]);
  red[threadIdx.x] = lm; __syncthreads();
  for (int off = 128; off >= 1; off >>= 1) {
    if (threadIdx.x < off) red[threadIdx.x] = fmaxf(red[threadIdx.x], red[threadIdx.x + off]);
    __syncthreads();
  }
  float m = red[0]; __syncthreads();
  float ls = 0.f;
  for (int s = threadIdx.x; s < S_; s += 256)
    ls += __expf(alpha[(size_t)(b * S_ + s) * 8 + hh] - m);
  red[threadIdx.x] = ls; __syncthreads();
  for (int off = 128; off >= 1; off >>= 1) {
    if (threadIdx.x < off) red[threadIdx.x] += red[threadIdx.x + off];
    __syncthreads();
  }
  if (threadIdx.x == 0) { mbuf[bh] = m; lbuf[bh] = red[0]; }
}

__global__ void pool_weights_k(const float* __restrict__ alpha,
                               const float* __restrict__ mbuf,
                               const float* __restrict__ lbuf,
                               float* __restrict__ wbuf) {
  int idx = blockIdx.x * 256 + threadIdx.x;   // B*S
  int b = idx >> 11;
  float w = 0.f;
#pragma unroll
  for (int hh = 0; hh < 8; ++hh)
    w += __expf(alpha[(size_t)idx * 8 + hh] - mbuf[b * 8 + hh]) / lbuf[b * 8 + hh];
  wbuf[idx] = w;
}

__global__ __launch_bounds__(512) void pool_out_k(const float* __restrict__ wbuf,
                                                  const _Float16* __restrict__ hbuf,
                                                  float* __restrict__ out) {
  const int b = blockIdx.x, d = threadIdx.x;
  float acc = 0.f;
  for (int s = 0; s < S_; ++s)
    acc += wbuf[b * S_ + s] * (float)hbuf[(size_t)(b * S_ + s) * D_ + d];
  out[b * D_ + d] = acc;
}

// ---------------------------------------------------------------------------
extern "C" void kernel_launch(void* const* d_in, const int* in_sizes, int n_in,
                              void* d_out, int out_size, void* d_ws, size_t ws_size,
                              hipStream_t stream) {
  (void)in_sizes; (void)n_in; (void)out_size; (void)ws_size;
  char* ws = (char*)d_ws;
  const size_t ACT = (size_t)B_ * S_ * D_;          // 4,194,304 f16 elems
  _Float16* s0 = (_Float16*)ws;
  _Float16* s1 = s0 + ACT;
  _Float16* s2 = s1 + ACT;
  _Float16* s3 = s2 + ACT;
  _Float16* s4 = s3 + ACT;
  _Float16* s5 = s4 + ACT;
  _Float16* vT = s5 + ACT;
  _Float16* wT = vT + ACT;              // 10 x 262144 f16
  _Float16* bh = wT + 10 * 262144;      // 10 x 512 f16
  float* alpha = (float*)(bh + 10 * 512);
  float* mbuf  = alpha + (size_t)B_ * S_ * 8;
  float* lbuf  = mbuf + 64;
  float* wbuf  = lbuf + 64;

  // weight order: q1 k1 v1 o1 q2 k2 v2 o2 ffn1 ffn2
  const int widx[10] = {2, 4, 6, 8, 10, 12, 14, 16, 20, 22};
  const int bidx[10] = {3, 5, 7, 9, 11, 13, 15, 17, 21, 23};
  for (int i = 0; i < 10; ++i) {
    wT_cvt<<<1024, 256, 0, stream>>>((const float*)d_in[widx[i]], wT + (size_t)i * 262144);
    b_cvt<<<2, 256, 0, stream>>>((const float*)d_in[bidx[i]], bh + (size_t)i * 512);
  }
  const float* x    = (const float*)d_in[0];
  const float* mask = (const float*)d_in[1];

  l2norm_cvt<<<1024, 256, 0, stream>>>(x, s0);

  dim3 gg(64, 4), gb(256);
  // ---- layer 1 ----
  gemm_f16<<<gg, gb, 0, stream>>>(s0, wT + 0 * 262144, bh + 0 * 512, s1, 0);  // Q
  gemm_f16<<<gg, gb, 0, stream>>>(s0, wT + 1 * 262144, bh + 1 * 512, s2, 0);  // K
  gemm_f16<<<gg, gb, 0, stream>>>(s0, wT + 2 * 262144, bh + 2 * 512, s3, 0);  // V
  head_transpose<<<16384, 256, 0, stream>>>(s3, vT);
  attn_fa<<<dim3(128, 8, 4), 32, 0, stream>>>(s1, s2, vT, mask, s4);
  gemm_f16<<<gg, gb, 0, stream>>>(s4, wT + 3 * 262144, bh + 3 * 512, s5, 0);  // O-proj
  layernorm_f16<<<1024, 256, 0, stream>>>(s5, (const float*)d_in[18], (const float*)d_in[19], s0);
  gemm_f16<<<gg, gb, 0, stream>>>(s0, wT + 8 * 262144, bh + 8 * 512, s1, 1);  // FFN1+ReLU
  gemm_f16<<<gg, gb, 0, stream>>>(s1, wT + 9 * 262144, bh + 9 * 512, s5, 0);  // FFN2
  // ---- layer 2 ----
  gemm_f16<<<gg, gb, 0, stream>>>(s5, wT + 4 * 262144, bh + 4 * 512, s0, 0);  // Q2
  gemm_f16<<<gg, gb, 0, stream>>>(s5, wT + 5 * 262144, bh + 5 * 512, s1, 0);  // K2
  gemm_f16<<<gg, gb, 0, stream>>>(s5, wT + 6 * 262144, bh + 6 * 512, s2, 0);  // V2
  head_transpose<<<16384, 256, 0, stream>>>(s2, vT);
  attn_fa<<<dim3(128, 8, 4), 32, 0, stream>>>(s0, s1, vT, mask, s3);
  gemm_f16<<<gg, gb, 0, stream>>>(s3, wT + 7 * 262144, bh + 7 * 512, s4, 0);  // O2-proj
  layernorm_f16<<<1024, 256, 0, stream>>>(s4, (const float*)d_in[24], (const float*)d_in[25], s5);
  // ---- pooling ----
  pool_alpha_k<<<1024, 256, 0, stream>>>(s5, (const float*)d_in[26], (const float*)d_in[27], mask, alpha);
  pool_stats_k<<<32, 256, 0, stream>>>(alpha, mbuf, lbuf);
  pool_weights_k<<<32, 256, 0, stream>>>(alpha, mbuf, lbuf, wbuf);
  pool_out_k<<<4, 512, 0, stream>>>(wbuf, s5, (float*)d_out);
}
